// ContextAttention_49349174231212
// MI455X (gfx1250) — compile-verified, matrix operand order checked by reading, not run on hardware
//
#include <hip/hip_runtime.h>

#define SEQ   1024
#define BATCH 4
#define EMB   1024
#define HEADS 16
#define DH    64

#define LDS_AS __attribute__((address_space(3)))

typedef __attribute__((ext_vector_type(16))) __bf16 v16bf;
typedef __attribute__((ext_vector_type(8)))  float  v8f;
typedef __attribute__((ext_vector_type(4)))  float  f32x4;
typedef LDS_AS float lds_f;

// Build a 16-element bf16 WMMA A/B fragment from four float4s using the
// native packed f32->bf16 convert path (v_cvt_pk_bf16_f32).
__device__ __forceinline__ v16bf frag_from_f4(f32x4 a, f32x4 b, f32x4 c, f32x4 d) {
  v16bf f;
  f[0]  = (__bf16)a[0]; f[1]  = (__bf16)a[1]; f[2]  = (__bf16)a[2]; f[3]  = (__bf16)a[3];
  f[4]  = (__bf16)b[0]; f[5]  = (__bf16)b[1]; f[6]  = (__bf16)b[2]; f[7]  = (__bf16)b[3];
  f[8]  = (__bf16)c[0]; f[9]  = (__bf16)c[1]; f[10] = (__bf16)c[2]; f[11] = (__bf16)c[3];
  f[12] = (__bf16)d[0]; f[13] = (__bf16)d[1]; f[14] = (__bf16)d[2]; f[15] = (__bf16)d[3];
  return f;
}

// Fragment from a K-contiguous f32 row in global memory:
// lanes 0-15 hold K = {kb..kb+7, 16+kb..16+kb+7} with kb=0, lanes 16-31 kb=8.
__device__ __forceinline__ v16bf load_frag_k_contig(const float* row, int kb) {
  f32x4 q0 = *(const f32x4*)(row + kb);
  f32x4 q1 = *(const f32x4*)(row + kb + 4);
  f32x4 q2 = *(const f32x4*)(row + 16 + kb);
  f32x4 q3 = *(const f32x4*)(row + 16 + kb + 4);
  return frag_from_f4(q0, q1, q2, q3);
}

// Same pattern but from LDS (addrspace(3) -> ds_load_b128).
__device__ __forceinline__ v16bf load_frag_lds(const lds_f* row, int kb) {
  f32x4 q0 = *(const LDS_AS f32x4*)(row + kb);
  f32x4 q1 = *(const LDS_AS f32x4*)(row + kb + 4);
  f32x4 q2 = *(const LDS_AS f32x4*)(row + 16 + kb);
  f32x4 q3 = *(const LDS_AS f32x4*)(row + 16 + kb + 4);
  return frag_from_f4(q0, q1, q2, q3);
}

// ---------------- g = masked mean over sequence -------------------------
__global__ void k_reduce_g(const float* __restrict__ x, const int* __restrict__ mask,
                           float* __restrict__ g) {
  int id = blockIdx.x * blockDim.x + threadIdx.x;   // B*EMB threads
  int b = id >> 10;
  int e = id & 1023;
  const float* xe = x + (size_t)b * EMB + e;        // x is (S,B,E)
  const int*   mb = mask + b * SEQ;
  float acc = 0.f, cnt = 0.f;
  for (int s = 0; s < SEQ; ++s) {
    float m = (float)mb[s];
    acc += m * xe[(size_t)s * (BATCH * EMB)];
    cnt += m;
  }
  g[id] = acc / fmaxf(cnt, 1e-9f);
}

// ---------------- gq/gk projections of g (tiny) -------------------------
__global__ void k_gproj(const float* __restrict__ Wgq, const float* __restrict__ bgq,
                        const float* __restrict__ Wgk, const float* __restrict__ bgk,
                        const float* __restrict__ g, float* __restrict__ gq, float* __restrict__ gk) {
  int id  = blockIdx.x * blockDim.x + threadIdx.x;  // 2*B*EMB threads
  int sel = id >> 12;                               // B*EMB = 4096
  int rem = id & 4095;
  int b = rem >> 10;
  int o = rem & 1023;
  const float* W    = sel ? Wgk : Wgq;
  const float* bias = sel ? bgk : bgq;
  const float* gb = g + b * EMB;
  const float* wr = W + (size_t)o * EMB;
  float acc = 0.f;
  for (int e = 0; e < EMB; ++e) acc += wr[e] * gb[e];
  (sel ? gk : gq)[rem] = acc + bias[o];
}

// ---------------- Q/K/V projection GEMM (WMMA bf16, async W staging) ----
// Block = 8 waves covering a 128(M) x 16(N) output strip; the 16xEMB weight
// rows are staged 32-K at a time into LDS with global_load_async_to_lds_b128
// and shared by all 8 waves via ds_load_b128.
__global__ void k_qkv_gemm(const float* __restrict__ x,
                           const float* __restrict__ Wq, const float* __restrict__ bq,
                           const float* __restrict__ Wk, const float* __restrict__ bk,
                           const float* __restrict__ Wv, const float* __restrict__ bv,
                           float* __restrict__ Q, float* __restrict__ K, float* __restrict__ V) {
  const float* W; const float* bias; float* O;
  if (blockIdx.z == 0)      { W = Wq; bias = bq; O = Q; }
  else if (blockIdx.z == 1) { W = Wk; bias = bk; O = K; }
  else                      { W = Wv; bias = bv; O = V; }

  __shared__ float wtile[16 * 32];                 // 2 KB f32 W slab

  // LDS-space view of the slab. Laundering the base pointer through an asm
  // makes the async-copy's "memory" clobber alias it, so the ds reads are
  // kept and re-issued every K step.
  lds_f* wl = (lds_f*)wtile;
  unsigned my_lds = (unsigned)(size_t)(wl + threadIdx.x * 4);   // 16B per thread
  asm volatile("" : "+v"(wl));

  int lane  = threadIdx.x & 31;
  int warp  = threadIdx.x >> 5;
  int tileN = blockIdx.x & 63;                    // N = EMB -> 64 tiles
  int tileM = (blockIdx.x >> 6) * 8 + warp;       // M = B*S -> 256 tiles
  int m  = lane & 15;
  int kb = (lane >> 4) << 3;

  int r = tileM * 16 + m;
  int b = r >> 10, t = r & 1023;
  const float* xrow = x + (size_t)t * (BATCH * EMB) + (size_t)b * EMB;   // x (S,B,E)

  // async staging: threads 0..127 each copy 16B of the W slab
  const float* gsrc = W + (size_t)(tileN * 16 + (threadIdx.x >> 3)) * EMB
                        + ((threadIdx.x & 7) << 2);
  bool stager = threadIdx.x < 128;                // waves 0-3

  v8f c = {};
  #pragma unroll 2
  for (int k0 = 0; k0 < EMB; k0 += 32) {
    if (stager) {
      asm volatile("global_load_async_to_lds_b128 %0, %1, off"
                   :: "v"(my_lds), "v"(gsrc + k0) : "memory");
    }
    asm volatile("s_wait_asynccnt 0" ::: "memory");
    __syncthreads();
    __builtin_prefetch(xrow + k0 + 32, 0, 3);
    v16bf a  = load_frag_k_contig(xrow + k0, kb);
    v16bf bm = load_frag_lds(wl + (lane & 15) * 32, kb);   // ds_load_b128
    c = __builtin_amdgcn_wmma_f32_16x16x32_bf16(false, a, false, bm, (short)0, c, false, false);
    __syncthreads();
  }
  int mo = (lane >> 4) << 3;
  int Oc = tileN * 16 + (lane & 15);
  float bval = bias[Oc];
  int h = Oc >> 6, d = Oc & 63;
  #pragma unroll
  for (int i = 0; i < 8; ++i) {
    int R = tileM * 16 + mo + i;
    int bb = R >> 10, tt = R & 1023;
    O[(((size_t)bb * HEADS + h) * SEQ + tt) * DH + d] = c[i] + bval;
  }
}

// ---------------- sigmoid gating, in-place on Q or K --------------------
__global__ void k_gate(float* __restrict__ Q, float* __restrict__ K,
                       const float* __restrict__ gq, const float* __restrict__ gk,
                       const float* __restrict__ qg_wq, const float* __restrict__ qg_bq,
                       const float* __restrict__ qg_wk, const float* __restrict__ qg_bk,
                       const float* __restrict__ kg_wq, const float* __restrict__ kg_bq,
                       const float* __restrict__ kg_wk, const float* __restrict__ kg_bk) {
  int idx = blockIdx.x * blockDim.x + threadIdx.x;  // B*H*S threads
  float* base; const float* gp; const float* wqv; const float* wkv; float b0, b1;
  if (blockIdx.z == 0) { base = Q; gp = gq; wqv = qg_wq; wkv = qg_wk; b0 = qg_bq[0]; b1 = qg_bk[0]; }
  else                 { base = K; gp = gk; wqv = kg_wq; wkv = kg_wk; b0 = kg_bq[0]; b1 = kg_bk[0]; }
  int b = idx >> 14;               // H*S = 16384
  int h = (idx >> 10) & 15;
  float* row = base + (size_t)idx * DH;
  const float* gv = gp + b * EMB + h * DH;
  float d1 = 0.f, d2 = 0.f;
  #pragma unroll 8
  for (int d = 0; d < DH; ++d) { d1 += row[d] * wqv[d]; d2 += gv[d] * wkv[d]; }
  float z = d1 + b0 + d2 + b1;
  float alpha = 1.f / (1.f + __expf(-z));
  #pragma unroll 8
  for (int d = 0; d < DH; ++d) row[d] = (1.f - alpha) * row[d] + alpha * gv[d];
}

// ---------------- scores = q_hat k_hat^T / sqrt(D), masked (WMMA) -------
__global__ void k_scores(const float* __restrict__ Q, const float* __restrict__ K,
                         const int* __restrict__ mask, float* __restrict__ P) {
  int lane = threadIdx.x & 31;
  int warp = threadIdx.x >> 5;
  int w    = blockIdx.x * 8 + warp;
  int bh   = w >> 12;              // 64*64 = 4096 tiles per (b,h)
  int tid  = w & 4095;
  int tileM = tid >> 6;
  int tileN = tid & 63;
  int m  = lane & 15;
  int kb = (lane >> 4) << 3;

  const float* qrow = Q + ((size_t)bh * SEQ + tileM * 16 + m) * DH;
  const float* krow = K + ((size_t)bh * SEQ + tileN * 16 + m) * DH;
  v8f c = {};
  #pragma unroll
  for (int k0 = 0; k0 < DH; k0 += 32) {
    v16bf a  = load_frag_k_contig(qrow + k0, kb);
    v16bf bm = load_frag_k_contig(krow + k0, kb);
    c = __builtin_amdgcn_wmma_f32_16x16x32_bf16(false, a, false, bm, (short)0, c, false, false);
  }
  int mo = (lane >> 4) << 3;
  int kc = tileN * 16 + (lane & 15);
  int b  = bh >> 4;
  #pragma unroll
  for (int i = 0; i < 8; ++i) {
    int qR = tileM * 16 + mo + i;
    float val = c[i] * 0.125f;                 // 1/sqrt(64)
    if (mask[b * SEQ + qR] == 0) val = -1.0e9f;
    P[((size_t)bh * SEQ + qR) * SEQ + kc] = val;
  }
}

// ---------------- softmax over last dim, in place on p_attn -------------
__global__ void k_softmax(float* __restrict__ P) {
  __shared__ float red[256];
  int t = threadIdx.x;
  float* row = P + (size_t)blockIdx.x * SEQ;
  f32x4 v = ((f32x4*)row)[t];
  red[t] = fmaxf(fmaxf(v[0], v[1]), fmaxf(v[2], v[3]));
  __syncthreads();
  for (int s = 128; s > 0; s >>= 1) { if (t < s) red[t] = fmaxf(red[t], red[t + s]); __syncthreads(); }
  float rowmax = red[0];
  __syncthreads();
  v[0] = __expf(v[0] - rowmax); v[1] = __expf(v[1] - rowmax);
  v[2] = __expf(v[2] - rowmax); v[3] = __expf(v[3] - rowmax);
  red[t] = v[0] + v[1] + v[2] + v[3];
  __syncthreads();
  for (int s = 128; s > 0; s >>= 1) { if (t < s) red[t] += red[t + s]; __syncthreads(); }
  float inv = 1.f / red[0];
  v *= inv;
  ((f32x4*)row)[t] = v;
}

// ---------------- ctx = p_attn @ v (WMMA, p converted on the fly) -------
__global__ void k_pv(const float* __restrict__ P, const float* __restrict__ V,
                     float* __restrict__ CTX) {
  int lane = threadIdx.x & 31;
  int warp = threadIdx.x >> 5;
  int w    = blockIdx.x * 8 + warp;
  int bh   = w >> 8;               // 64*4 = 256 tiles per (b,h)
  int tid  = w & 255;
  int tileM = tid >> 2;
  int tileN = tid & 3;
  int m  = lane & 15;
  int kb = (lane >> 4) << 3;

  const float* prow  = P + ((size_t)bh * SEQ + tileM * 16 + m) * SEQ;
  int dcol = tileN * 16 + m;       // B-fragment column for this lane
  const float* vbase = V + (size_t)bh * SEQ * DH + dcol;

  v8f c = {};
  #pragma unroll 4
  for (int k0 = 0; k0 < SEQ; k0 += 32) {
    __builtin_prefetch(prow + k0 + 32, 0, 3);
    v16bf a = load_frag_k_contig(prow + k0, kb);
    v16bf bm;
    const float* vb = vbase + (size_t)k0 * DH;
    #pragma unroll
    for (int p = 0; p < 8; ++p) bm[p]     = (__bf16)vb[(size_t)(kb + p) * DH];
    #pragma unroll
    for (int p = 0; p < 8; ++p) bm[8 + p] = (__bf16)vb[(size_t)(16 + kb + p) * DH];
    c = __builtin_amdgcn_wmma_f32_16x16x32_bf16(false, a, false, bm, (short)0, c, false, false);
  }
  int mo = (lane >> 4) << 3;
  int n  = lane & 15;
  int b = bh >> 4, h = bh & 15;
  int dout = tileN * 16 + n;
  #pragma unroll
  for (int i = 0; i < 8; ++i) {
    int qR = tileM * 16 + mo + i;
    CTX[((size_t)b * SEQ + qR) * EMB + h * DH + dout] = c[i];
  }
}

// ---------------- output projection (WMMA, async W staging) -------------
__global__ void k_out_gemm(const float* __restrict__ CTX, const float* __restrict__ Wo,
                           const float* __restrict__ bo, float* __restrict__ out) {
  __shared__ float wtile[16 * 32];

  lds_f* wl = (lds_f*)wtile;
  unsigned my_lds = (unsigned)(size_t)(wl + threadIdx.x * 4);
  asm volatile("" : "+v"(wl));

  int lane  = threadIdx.x & 31;
  int warp  = threadIdx.x >> 5;
  int tileN = blockIdx.x & 63;
  int tileM = (blockIdx.x >> 6) * 8 + warp;
  int m  = lane & 15;
  int kb = (lane >> 4) << 3;

  const float* crow = CTX + (size_t)(tileM * 16 + m) * EMB;   // rows are (b*S+s)

  const float* gsrc = Wo + (size_t)(tileN * 16 + (threadIdx.x >> 3)) * EMB
                         + ((threadIdx.x & 7) << 2);
  bool stager = threadIdx.x < 128;

  v8f c = {};
  #pragma unroll 2
  for (int k0 = 0; k0 < EMB; k0 += 32) {
    if (stager) {
      asm volatile("global_load_async_to_lds_b128 %0, %1, off"
                   :: "v"(my_lds), "v"(gsrc + k0) : "memory");
    }
    asm volatile("s_wait_asynccnt 0" ::: "memory");
    __syncthreads();
    __builtin_prefetch(crow + k0 + 32, 0, 3);
    v16bf a  = load_frag_k_contig(crow + k0, kb);
    v16bf bm = load_frag_lds(wl + (lane & 15) * 32, kb);
    c = __builtin_amdgcn_wmma_f32_16x16x32_bf16(false, a, false, bm, (short)0, c, false, false);
    __syncthreads();
  }
  int mo = (lane >> 4) << 3;
  int Oc = tileN * 16 + (lane & 15);
  float bval = bo[Oc];
  #pragma unroll
  for (int i = 0; i < 8; ++i) {
    int R = tileM * 16 + mo + i;
    int bb = R >> 10, ss = R & 1023;
    out[((size_t)ss * BATCH + bb) * EMB + Oc] = c[i] + bval;    // out is (S,B,E)
  }
}

extern "C" void kernel_launch(void* const* d_in, const int* in_sizes, int n_in,
                              void* d_out, int out_size, void* d_ws, size_t ws_size,
                              hipStream_t stream) {
  (void)in_sizes; (void)n_in; (void)out_size; (void)ws_size;
  const float* x     = (const float*)d_in[0];
  const int*   mask  = (const int*)  d_in[1];
  const float* Wq    = (const float*)d_in[2];
  const float* bq    = (const float*)d_in[3];
  const float* Wk    = (const float*)d_in[4];
  const float* bk    = (const float*)d_in[5];
  const float* Wv    = (const float*)d_in[6];
  const float* bv    = (const float*)d_in[7];
  const float* Wo    = (const float*)d_in[8];
  const float* bo    = (const float*)d_in[9];
  const float* Wgq   = (const float*)d_in[10];
  const float* bgq   = (const float*)d_in[11];
  const float* Wgk   = (const float*)d_in[12];
  const float* bgk   = (const float*)d_in[13];
  const float* qg_wq = (const float*)d_in[14];
  const float* qg_bq = (const float*)d_in[15];
  const float* qg_wk = (const float*)d_in[16];
  const float* qg_bk = (const float*)d_in[17];
  const float* kg_wq = (const float*)d_in[18];
  const float* kg_bq = (const float*)d_in[19];
  const float* kg_wk = (const float*)d_in[20];
  const float* kg_bk = (const float*)d_in[21];

  float* out = (float*)d_out;                             // (S,B,E)
  float* P   = out + (size_t)SEQ * BATCH * EMB;           // p_attn (B,H,S,S)

  const size_t QKV = (size_t)BATCH * HEADS * SEQ * DH;    // 4,194,304 floats
  float* ws  = (float*)d_ws;
  float* Q   = ws;
  float* K   = ws + QKV;
  float* V   = ws + 2 * QKV;
  float* CTX = ws + 3 * QKV;                              // (B,S,E)
  float* g   = ws + 4 * QKV;                              // (B,E)
  float* gq  = g  + BATCH * EMB;
  float* gk  = gq + BATCH * EMB;

  k_reduce_g<<<dim3(16), dim3(256), 0, stream>>>(x, mask, g);
  k_gproj  <<<dim3(32), dim3(256), 0, stream>>>(Wgq, bgq, Wgk, bgk, g, gq, gk);
  k_qkv_gemm<<<dim3(2048, 1, 3), dim3(256), 0, stream>>>(x, Wq, bq, Wk, bk, Wv, bv, Q, K, V);
  k_gate   <<<dim3(256, 1, 2), dim3(256), 0, stream>>>(Q, K, gq, gk,
                                                       qg_wq, qg_bq, qg_wk, qg_bk,
                                                       kg_wq, kg_bq, kg_wk, kg_bk);
  k_scores <<<dim3(32768), dim3(256), 0, stream>>>(Q, K, mask, P);
  k_softmax<<<dim3(65536), dim3(256), 0, stream>>>(P);
  k_pv     <<<dim3(2048), dim3(256), 0, stream>>>(P, V, CTX);
  k_out_gemm<<<dim3(2048), dim3(256), 0, stream>>>(CTX, Wo, bo, out);
}
